// HRViTAttention_1211180778309
// MI455X (gfx1250) — matrix operand order repaired and checked by reading, not compiled, not run on hardware
//
#include <hip/hip_runtime.h>
#include <hip/hip_bf16.h>

typedef __attribute__((ext_vector_type(16))) __bf16 bf16x16;
typedef __attribute__((ext_vector_type(8)))  __bf16 bf16x8;
typedef __attribute__((ext_vector_type(8)))  float  floatx8;
typedef __attribute__((ext_vector_type(4)))  float  floatx4;

#define SHUF16(lo, hi) __builtin_shufflevector((lo), (hi), 0,1,2,3,4,5,6,7,8,9,10,11,12,13,14,15)

// ---------------- constants (match reference) ----------------
#define BB   8
#define HH   64
#define WW   64
#define NN   (HH * WW)          // 4096
#define DIMC 256
#define ROWS (BB * NN)          // 32768

// workspace layout (float offsets)
constexpr size_t OFF_QV    = 0;          // [ROWS, 512]
constexpr size_t OFF_VCONV = 16777216;   // [ROWS, 256]
constexpr size_t OFF_OH    = 25165824;   // [ROWS, 128]
constexpr size_t OFF_OV    = 29360128;   // [ROWS, 128]
constexpr size_t OFF_CH    = 33554432;   // [ROWS, 128]
constexpr size_t OFF_CV    = 37748736;   // [ROWS, 128]
constexpr size_t OFF_FCONV = 41943040;   // [B, 144, H, W]
constexpr size_t OFF_A     = 46661632;   // [ROWS, 256]
constexpr size_t OFF_APROJ = 0;          // reuse qv region
constexpr size_t OFF_STATS = 55050240;   // [512]

__device__ __forceinline__ float hswish(float x) {
  return x * fminf(fmaxf(x + 3.f, 0.f), 6.f) * (1.f / 6.f);
}

__device__ __forceinline__ bf16x8 cvt8(floatx4 a, floatx4 b) {
  bf16x8 r;
  r[0] = (__bf16)a[0]; r[1] = (__bf16)a[1]; r[2] = (__bf16)a[2]; r[3] = (__bf16)a[3];
  r[4] = (__bf16)b[0]; r[5] = (__bf16)b[1]; r[6] = (__bf16)b[2]; r[7] = (__bf16)b[3];
  return r;
}

// ---------------- WMMA GEMM: C = A(MxK) * W(KxN) + bias ----------------
// block = 256 threads (8 waves), tile 128x64, K-step 32.
// As row-major (A-frag = two contiguous b128 per lane);
// Bt transposed [n][k] (B-frag = two contiguous b128 per lane).
__global__ __launch_bounds__(256) void gemm_bias_wmma(
    const float* __restrict__ A, const float* __restrict__ W,
    const float* __restrict__ bias, float* __restrict__ C,
    int M, int K, int Nc)
{
  __shared__ __align__(16) __bf16 As[128][32];
  __shared__ __align__(16) __bf16 Bt[64][32];
  const int tid = threadIdx.x;
  const int lane = tid & 31, wave = tid >> 5;
  const int laneLo = lane & 15, laneHi = lane >> 4;
  const int bm = blockIdx.x * 128;
  const int bn = blockIdx.y * 64;
  const int mRow = wave * 16;              // wave's 16 rows in tile (all 64 cols)

  floatx8 acc[4] = {};

  for (int k0 = 0; k0 < K; k0 += 32) {
    // A tile: 128x32, each thread converts two 8-elem chunks (2x global b128 -> 1 LDS b128)
    for (int i = tid; i < 128 * 4; i += 256) {
      int r = i >> 2, c8 = (i & 3) * 8;
      const float* src = &A[(size_t)(bm + r) * K + k0 + c8];
      *(bf16x8*)&As[r][c8] = cvt8(*(const floatx4*)src, *(const floatx4*)(src + 4));
    }
    // W tile transposed: Bt[n][k]
    for (int i = tid; i < 32 * 16; i += 256) {
      int r = i >> 4, c4 = (i & 15) * 4;
      floatx4 w4 = *(const floatx4*)&W[(size_t)(k0 + r) * Nc + bn + c4];
      Bt[c4 + 0][r] = (__bf16)w4[0];
      Bt[c4 + 1][r] = (__bf16)w4[1];
      Bt[c4 + 2][r] = (__bf16)w4[2];
      Bt[c4 + 3][r] = (__bf16)w4[3];
    }
    __syncthreads();

    // A fragment: lane row mRow+laneLo, K chunks {laneHi*8..+8, 16+laneHi*8..+8}
    bf16x8 alo = *(const bf16x8*)&As[mRow + laneLo][laneHi * 8];
    bf16x8 ahi = *(const bf16x8*)&As[mRow + laneLo][16 + laneHi * 8];
    bf16x16 af = SHUF16(alo, ahi);
#pragma unroll
    for (int t = 0; t < 4; ++t) {
      // B fragment: lane col t*16+laneLo, K chunks {laneHi*8..+8, 16+laneHi*8..+8}
      bf16x8 blo = *(const bf16x8*)&Bt[t * 16 + laneLo][laneHi * 8];
      bf16x8 bhi = *(const bf16x8*)&Bt[t * 16 + laneLo][16 + laneHi * 8];
      bf16x16 bfr = SHUF16(blo, bhi);
      acc[t] = __builtin_amdgcn_wmma_f32_16x16x32_bf16(false, af, false, bfr, (short)0, acc[t], false, false);
    }
    __syncthreads();
  }
  // C layout: elem v -> row mRow + v + laneHi*8, col t*16 + laneLo
#pragma unroll
  for (int t = 0; t < 4; ++t) {
    int c0 = bn + t * 16 + laneLo;
    float bv = bias[c0];
#pragma unroll
    for (int v = 0; v < 8; ++v) {
      int gr = bm + mRow + v + laneHi * 8;
      C[(size_t)gr * Nc + c0] = acc[t][v] + bv;
    }
  }
}

// ---------------- hard_swish + depthwise 3x3 on v (groups=256) ----------------
__global__ void vconv_dw(const float* __restrict__ qv, const float* __restrict__ w,
                         const float* __restrict__ bias, float* __restrict__ out)
{
  int idx = blockIdx.x * blockDim.x + threadIdx.x;   // (b*N+n)*256 + c
  int c = idx & 255;
  int n = (idx >> 8) & (NN - 1);
  int b = idx >> 20;
  int y = n >> 6, x = n & 63;
  float acc = bias[c];
#pragma unroll
  for (int ky = 0; ky < 3; ++ky) {
    int yy = y + ky - 1;
    if (yy < 0 || yy >= HH) continue;
#pragma unroll
    for (int kx = 0; kx < 3; ++kx) {
      int xx = x + kx - 1;
      if (xx < 0 || xx >= WW) continue;
      float v = qv[((size_t)(b * NN + yy * WW + xx)) * 512 + 256 + c];
      acc += w[c * 9 + ky * 3 + kx] * hswish(v);
    }
  }
  out[idx] = acc;
}

// ---------------- conv-attention branch: build f_conv [B,144,H,W] ----------------
__global__ void fconv_build(const float* __restrict__ qv, int qOff, int kvOff,
                            const float* __restrict__ fc_w, float* __restrict__ fconv)
{
  int idx = blockIdx.x * blockDim.x + threadIdx.x;   // b*144*N + ch*N + n
  int n = idx & (NN - 1);
  int ch = (idx >> 12) % 144;
  int b = idx / (144 * NN);
  int s = ch / 9, o = ch % 9;
  const float* row = qv + (size_t)(b * NN + n) * 512;
  float acc = 0.f;
#pragma unroll
  for (int h = 0; h < 8; ++h) {
    float qe = row[qOff + h * 16 + s];
    float ke = row[kvOff + h * 16 + s];
    acc += fc_w[o * 24 + h] * qe + (fc_w[o * 24 + 8 + h] + fc_w[o * 24 + 16 + h]) * ke;
  }
  fconv[idx] = acc;
}

// grouped conv (groups=16): out[b,n,oc], oc<128, 9 in-ch per group, 3x3, pad 1
__global__ void dwgroup_conv(const float* __restrict__ fconv, const float* __restrict__ dep_w,
                             const float* __restrict__ dep_b, float* __restrict__ out)
{
  int idx = blockIdx.x * blockDim.x + threadIdx.x;   // (b*N+n)*128 + oc
  int oc = idx & 127;
  int n = (idx >> 7) & (NN - 1);
  int b = idx >> 19;
  int y = n >> 6, x = n & 63;
  int gch = (oc >> 3) * 9;
  float acc = dep_b[oc];
  for (int j = 0; j < 9; ++j) {
    const float* w = dep_w + ((size_t)oc * 9 + j) * 9;
    const float* fc = fconv + ((size_t)b * 144 + gch + j) * NN;
#pragma unroll
    for (int ky = 0; ky < 3; ++ky) {
      int yy = y + ky - 1;
      if (yy < 0 || yy >= HH) continue;
#pragma unroll
      for (int kx = 0; kx < 3; ++kx) {
        int xx = x + kx - 1;
        if (xx < 0 || xx >= WW) continue;
        acc += w[ky * 3 + kx] * fc[yy * WW + xx];
      }
    }
  }
  out[idx] = acc;
}

// ---------------- flash-style cross-shaped window attention (WMMA) ----------------
// one block = (batch b, window g, head h, q-tile of 128 rows); L=512, dh=32, K==V
__global__ __launch_bounds__(256) void flash_attn_wmma(
    const float* __restrict__ qv, int qOff, int kvOff,
    float* __restrict__ outp, int horizontal)
{
  __shared__ __align__(16) __bf16 Qs[128][32];   // q rows x dh
  __shared__ __align__(16) __bf16 Ks[64][32];    // key x dh   (QK^T B-operand)
  __shared__ __align__(16) __bf16 Kt[32][64];    // dh x key   (PV   B-operand)
  __shared__ __align__(16) __bf16 Ps[8][16][64]; // per-wave P
  const int tid = threadIdx.x;
  const int lane = tid & 31, wave = tid >> 5;
  const int laneLo = lane & 15, laneHi = lane >> 4;
  const int bx = blockIdx.x;
  const int qt = bx & 3;
  const int h  = (bx >> 2) & 3;
  const int g  = (bx >> 4) & 7;
  const int b  = bx >> 7;
  const int qT0 = qt * 128;
  const float scale = 0.17677669529663687f;   // 32^-0.5

  auto mapn = [&](int t) -> int {
    if (horizontal) return g * 512 + t;       // n = (g*ws+r)*W + c, contiguous
    int s = t >> 6, y = t & 63;               // t = s*H + y
    return y * WW + g * 8 + s;
  };
  const float* base = qv + (size_t)b * NN * 512;

  for (int i = tid; i < 128 * 4; i += 256) {
    int r = i >> 2, d8 = (i & 3) * 8;
    const float* src = &base[(size_t)mapn(qT0 + r) * 512 + qOff + h * 32 + d8];
    *(bf16x8*)&Qs[r][d8] = cvt8(*(const floatx4*)src, *(const floatx4*)(src + 4));
  }
  __syncthreads();

  // Q A-fragment (dh=32 => exactly one 16x16x32 per 16-key subtile)
  bf16x8 qlo = *(const bf16x8*)&Qs[wave * 16 + laneLo][laneHi * 8];
  bf16x8 qhi = *(const bf16x8*)&Qs[wave * 16 + laneLo][16 + laneHi * 8];
  bf16x16 qf = SHUF16(qlo, qhi);

  float mrow[8], lrow[8];
#pragma unroll
  for (int v = 0; v < 8; ++v) { mrow[v] = -1e30f; lrow[v] = 0.f; }
  floatx8 o0 = {}, o1 = {};

  for (int kc = 0; kc < 512; kc += 64) {
    __syncthreads();
    for (int i = tid; i < 64 * 4; i += 256) {
      int r = i >> 2, d8 = (i & 3) * 8;
      const float* src = &base[(size_t)mapn(kc + r) * 512 + kvOff + h * 32 + d8];
      bf16x8 v8 = cvt8(*(const floatx4*)src, *(const floatx4*)(src + 4));
      *(bf16x8*)&Ks[r][d8] = v8;
#pragma unroll
      for (int j = 0; j < 8; ++j) Kt[d8 + j][r] = v8[j];
    }
    __syncthreads();

    // S = Q * K^T : B-frag lane=key (row of Ks), K=d chunks {laneHi*8..+8, 16+laneHi*8..+8}
    floatx8 s[4];
#pragma unroll
    for (int t = 0; t < 4; ++t) {
      bf16x8 klo = *(const bf16x8*)&Ks[t * 16 + laneLo][laneHi * 8];
      bf16x8 khi = *(const bf16x8*)&Ks[t * 16 + laneLo][16 + laneHi * 8];
      bf16x16 bfr = SHUF16(klo, khi);
      floatx8 z = {};
      s[t] = __builtin_amdgcn_wmma_f32_16x16x32_bf16(false, qf, false, bfr, (short)0, z, false, false);
    }

    // online softmax per row (row = v + laneHi*8 in wave's 16-row tile)
#pragma unroll
    for (int v = 0; v < 8; ++v) {
      float mx = fmaxf(fmaxf(s[0][v] * scale, s[1][v] * scale),
                       fmaxf(s[2][v] * scale, s[3][v] * scale));
#pragma unroll
      for (int msk = 1; msk < 16; msk <<= 1)
        mx = fmaxf(mx, __shfl_xor(mx, msk, 32));
      float mn = fmaxf(mrow[v], mx);
      float alpha = __expf(mrow[v] - mn);
      mrow[v] = mn;
      float rs = 0.f;
#pragma unroll
      for (int t = 0; t < 4; ++t) {
        float pv = __expf(s[t][v] * scale - mn);
        s[t][v] = pv;
        rs += pv;
      }
#pragma unroll
      for (int msk = 1; msk < 16; msk <<= 1)
        rs += __shfl_xor(rs, msk, 32);
      lrow[v] = alpha * lrow[v] + rs;
      o0[v] *= alpha;
      o1[v] *= alpha;
    }
    // repack P through LDS into A-fragment layout
#pragma unroll
    for (int t = 0; t < 4; ++t)
#pragma unroll
      for (int v = 0; v < 8; ++v)
        Ps[wave][v + laneHi * 8][t * 16 + laneLo] = (__bf16)s[t][v];
    __syncthreads();

    // O += P * V  (P A-frags and V B-frags are contiguous b128 chunks)
#pragma unroll
    for (int half = 0; half < 2; ++half) {
      bf16x8 plo = *(const bf16x8*)&Ps[wave][laneLo][half * 32 + laneHi * 8];
      bf16x8 phi = *(const bf16x8*)&Ps[wave][laneLo][half * 32 + 16 + laneHi * 8];
      bf16x16 pf = SHUF16(plo, phi);

      bf16x8 v0lo = *(const bf16x8*)&Kt[laneLo][half * 32 + laneHi * 8];
      bf16x8 v0hi = *(const bf16x8*)&Kt[laneLo][half * 32 + 16 + laneHi * 8];
      bf16x16 vf0 = SHUF16(v0lo, v0hi);
      bf16x8 v1lo = *(const bf16x8*)&Kt[16 + laneLo][half * 32 + laneHi * 8];
      bf16x8 v1hi = *(const bf16x8*)&Kt[16 + laneLo][half * 32 + 16 + laneHi * 8];
      bf16x16 vf1 = SHUF16(v1lo, v1hi);

      o0 = __builtin_amdgcn_wmma_f32_16x16x32_bf16(false, pf, false, vf0, (short)0, o0, false, false);
      o1 = __builtin_amdgcn_wmma_f32_16x16x32_bf16(false, pf, false, vf1, (short)0, o1, false, false);
    }
  }

#pragma unroll
  for (int v = 0; v < 8; ++v) {
    int r = wave * 16 + v + laneHi * 8;
    int n = mapn(qT0 + r);
    float inv = 1.f / lrow[v];
    size_t off = ((size_t)(b * NN + n)) * 128 + h * 32;
    outp[off + laneLo]      = o0[v] * inv;
    outp[off + 16 + laneLo] = o1[v] * inv;
  }
}

// ---------------- combine: a = hswish(rate1*o + rate2*conv | concat + v_conv) ----------------
__global__ void combine_k(const float* __restrict__ oh, const float* __restrict__ ov,
                          const float* __restrict__ ch, const float* __restrict__ cv,
                          const float* __restrict__ vconv,
                          const float* __restrict__ r1p, const float* __restrict__ r2p,
                          float* __restrict__ a)
{
  int idx = blockIdx.x * blockDim.x + threadIdx.x;   // (b*N+n)*256 + c
  int c = idx & 255;
  size_t half = (size_t)(idx >> 8) * 128;
  float r1 = r1p[0], r2 = r2p[0];
  float t = (c < 128) ? (r1 * oh[half + c] + r2 * ch[half + c])
                      : (r1 * ov[half + c - 128] + r2 * cv[half + c - 128]);
  a[idx] = hswish(t + vconv[idx]);
}

// ---------------- BatchNorm (deterministic two-stage) ----------------
__global__ __launch_bounds__(256) void bn_stats(const float* __restrict__ x, float* __restrict__ stats)
{
  __shared__ float s1[256], s2[256];
  int ch = blockIdx.x, tid = threadIdx.x;
  float sum = 0.f, sq = 0.f;
  for (int r = tid; r < ROWS; r += 256) {
    float v = x[(size_t)r * DIMC + ch];
    sum += v; sq += v * v;
  }
  s1[tid] = sum; s2[tid] = sq;
  __syncthreads();
  for (int s = 128; s > 0; s >>= 1) {
    if (tid < s) { s1[tid] += s1[tid + s]; s2[tid] += s2[tid + s]; }
    __syncthreads();
  }
  if (tid == 0) {
    float mu = s1[0] * (1.f / ROWS);
    stats[ch] = mu;
    stats[DIMC + ch] = s2[0] * (1.f / ROWS) - mu * mu;
  }
}

__global__ void bn_apply(const float* __restrict__ x, const float* __restrict__ stats,
                         const float* __restrict__ g, const float* __restrict__ bb,
                         float* __restrict__ out)
{
  int idx = blockIdx.x * blockDim.x + threadIdx.x;
  int c = idx & 255;
  out[idx] = (x[idx] - stats[c]) * __frsqrt_rn(stats[DIMC + c] + 1e-5f) * g[c] + bb[c];
}

// ---------------- launch ----------------
extern "C" void kernel_launch(void* const* d_in, const int* in_sizes, int n_in,
                              void* d_out, int out_size, void* d_ws, size_t ws_size,
                              hipStream_t stream) {
  (void)in_sizes; (void)n_in; (void)out_size; (void)ws_size;
  const float* x       = (const float*)d_in[0];
  const float* w_qkv   = (const float*)d_in[1];
  const float* b_qkv   = (const float*)d_in[2];
  const float* w_out   = (const float*)d_in[3];
  const float* b_out   = (const float*)d_in[4];
  const float* bn_g    = (const float*)d_in[5];
  const float* bn_b    = (const float*)d_in[6];
  const float* pconv_w = (const float*)d_in[7];
  const float* pconv_b = (const float*)d_in[8];
  const float* fc_w    = (const float*)d_in[9];
  const float* dep_w   = (const float*)d_in[10];
  const float* dep_b   = (const float*)d_in[11];
  const float* rate1   = (const float*)d_in[12];
  const float* rate2   = (const float*)d_in[13];

  float* ws    = (float*)d_ws;
  float* qv    = ws + OFF_QV;
  float* vcv   = ws + OFF_VCONV;
  float* o_h   = ws + OFF_OH;
  float* o_v   = ws + OFF_OV;
  float* c_h   = ws + OFF_CH;
  float* c_v   = ws + OFF_CV;
  float* fcv   = ws + OFF_FCONV;
  float* a     = ws + OFF_A;
  float* aproj = ws + OFF_APROJ;
  float* stats = ws + OFF_STATS;
  float* out   = (float*)d_out;

  // 1) qv = x @ w_qkv + b_qkv   [32768 x 512]
  gemm_bias_wmma<<<dim3(ROWS / 128, 512 / 64), 256, 0, stream>>>(x, w_qkv, b_qkv, qv, ROWS, 256, 512);
  // 2) v branch: hswish + dw3x3
  vconv_dw<<<(ROWS * 256) / 256, 256, 0, stream>>>(qv, pconv_w, pconv_b, vcv);
  // 3) horizontal branch (qh: off 0, vh: off 256)
  fconv_build<<<(BB * 144 * NN) / 256, 256, 0, stream>>>(qv, 0, 256, fc_w, fcv);
  dwgroup_conv<<<(ROWS * 128) / 256, 256, 0, stream>>>(fcv, dep_w, dep_b, c_h);
  flash_attn_wmma<<<1024, 256, 0, stream>>>(qv, 0, 256, o_h, 1);
  // 4) vertical branch (qv2: off 128, vv: off 384)
  fconv_build<<<(BB * 144 * NN) / 256, 256, 0, stream>>>(qv, 128, 384, fc_w, fcv);
  dwgroup_conv<<<(ROWS * 128) / 256, 256, 0, stream>>>(fcv, dep_w, dep_b, c_v);
  flash_attn_wmma<<<1024, 256, 0, stream>>>(qv, 128, 384, o_v, 0);
  // 5) combine + hswish
  combine_k<<<(ROWS * 256) / 256, 256, 0, stream>>>(o_h, o_v, c_h, c_v, vcv, rate1, rate2, a);
  // 6) out projection  [32768 x 256]
  gemm_bias_wmma<<<dim3(ROWS / 128, 256 / 64), 256, 0, stream>>>(a, w_out, b_out, aproj, ROWS, 256, 256);
  // 7) batchnorm
  bn_stats<<<256, 256, 0, stream>>>(aproj, stats);
  bn_apply<<<(ROWS * 256) / 256, 256, 0, stream>>>(aproj, stats, bn_g, bn_b, out);
}
